// GraphSAGETaskDistributor_3762391351459
// MI455X (gfx1250) — matrix operand order repaired and checked by reading, not compiled, not run on hardware
//
#include <hip/hip_runtime.h>
#include <stdint.h>

// ---------------------------------------------------------------------------
// GraphSAGE 3-layer forward, MI455X (gfx1250, wave32, WMMA bf16->f32)
//   layer: out = lin_l(mean_j x_j) + lin_r(x_i)  [+ LayerNorm + ReLU]
//
// v2 changes vs v1:
//  - weights pre-packed once per launch into bf16 WMMA B-fragment layout
//    (2x global_load_b128 per fragment instead of 16 strided b32 loads)
//  - A tiles (agg*invcnt and x) staged cooperatively in LDS in the exact
//    bf16 A-fragment layout: one conversion per element per block instead of
//    one per wave; fragments fetched with 2x ds_load_b128
// ---------------------------------------------------------------------------

typedef __attribute__((ext_vector_type(16))) __bf16 v16bf;
typedef __attribute__((ext_vector_type(8)))  float  v8f;

#define NNODES 40000
#define NEDGES 640000
#define DH     128
#define DOUT3  64
#define LN_EPS 1e-5f

static_assert(NNODES % 16 == 0, "row tiles must be exact");

// pack two f32 into one dword of bf16 pair -> lowers to v_cvt_pk_bf16_f32
__device__ __forceinline__ unsigned int pack2bf(float a, float b) {
  union { __bf16 h[2]; unsigned int u; } p;
  p.h[0] = (__bf16)a;
  p.h[1] = (__bf16)b;
  return p.u;
}

// ---------------------------------------------------------------- zero (f4)
__global__ void k_zero4(float4* __restrict__ p, int n4) {
  int i = blockIdx.x * blockDim.x + threadIdx.x;
  if (i < n4) p[i] = make_float4(0.f, 0.f, 0.f, 0.f);
}

// ---------------------------------------------------------------- degrees
__global__ void k_degree(const long long* __restrict__ dst,
                         float* __restrict__ cnt, int ne) {
  int e = blockIdx.x * blockDim.x + threadIdx.x;
  if (e < ne) atomicAdd(&cnt[(int)dst[e]], 1.0f);
}

// ------------------------------------------------- scatter-add, wave/edge
// one wave32 per edge: lane handles 4 consecutive features (float4 load,
// 4x global_atomic_add_f32). Node features are L2-resident (20.5MB << 192MB).
__global__ void k_scatter(const float* __restrict__ x,
                          const long long* __restrict__ src,
                          const long long* __restrict__ dst,
                          float* __restrict__ agg, int ne) {
  int gid  = blockIdx.x * blockDim.x + threadIdx.x;
  int wave = gid >> 5;
  int lane = threadIdx.x & 31;
  if (wave >= ne) return;
  int s = (int)src[wave];
  int d = (int)dst[wave];
  const float4 v = *(const float4*)(x + (size_t)s * DH + lane * 4);
  float* o = agg + (size_t)d * DH + lane * 4;
  atomicAdd(o + 0, v.x);
  atomicAdd(o + 1, v.y);
  atomicAdd(o + 2, v.z);
  atomicAdd(o + 3, v.w);
}

// -------------------------------------------- weight pre-pack (once/launch)
// B-fragment layout chosen here (contiguous K, as A) for v_wmma_f32_16x16x32_bf16 (B is 32x16, K x N):
//   lane = column within 16-wide tile (+16 lanes hold K+16..K+31)
//   16 bf16 per lane = K run of 16. Fragment id = (coltile*4 + ktile).
// Wp[frag][lane][16 bf16]  -> 32B per lane, two b128 loads in the GEMM.
template <int OUTD>
__global__ void k_pack_w(const float* __restrict__ W,
                         unsigned int* __restrict__ Wp) {
  constexpr int NT = (OUTD / 16) * 4 * 32;  // threads = frags * lanes
  int t = blockIdx.x * blockDim.x + threadIdx.x;
  if (t >= NT) return;
  const int lane  = t & 31;
  const int kt    = (t >> 5) & 3;
  const int ct    = t >> 7;
  const int col   = ct * 16 + (lane & 15);
  const int kbase = kt * 32 + (lane >> 4) * 16;
  unsigned int o[8];
#pragma unroll
  for (int p = 0; p < 8; ++p)
    o[p] = pack2bf(W[(kbase + 2 * p) * OUTD + col],
                   W[(kbase + 2 * p + 1) * OUTD + col]);
  uint4* dst = (uint4*)(Wp + (size_t)t * 8);
  dst[0] = *(uint4*)&o[0];
  dst[1] = *(uint4*)&o[4];
}

// -------------------------------------------------------- fused SAGE GEMM
// Block: (OUTD/16) waves; wave w owns rows[16] x cols[w*16..+16].
// Stage 1: block packs agg*invcnt and x tiles into LDS in A-fragment layout.
// Stage 2: per ktile: 2x ds_load_b128 (A) + 2x global b128 (B) per matrix,
//          2 WMMA; acc starts at bias.
// Epilogue: direct store, or LDS LayerNorm + ReLU across the 128-wide row.
template <int OUTD, bool DO_LN>
__global__ __launch_bounds__((OUTD / 16) * 32)
void k_sage_gemm(const float* __restrict__ agg, const float* __restrict__ cnt,
                 const float* __restrict__ xin,
                 const uint4* __restrict__ Wlp, const uint4* __restrict__ Wrp,
                 const float* __restrict__ bias,
                 const float* __restrict__ gamma, const float* __restrict__ beta,
                 float* __restrict__ out) {
  // A-fragment staging: aS[ktile][lane][8 dwords] = 16 bf16 per lane
  __shared__ unsigned int aS[4][32][8];
  __shared__ unsigned int xS[4][32][8];
  // LN staging
  __shared__ float tile[16][DH + 4];
  __shared__ float psum[16][16];
  __shared__ float psq[16][16];
  __shared__ float mu_s[16];
  __shared__ float rs_s[16];

  const int tid   = threadIdx.x;
  const int wave  = tid >> 5;
  const int lane  = tid & 31;
  const int half  = lane >> 4;
  const int m     = lane & 15;
  const int rbase = blockIdx.x * 16;
  const int col   = wave * 16 + m;

  // ---- Stage 1: cooperative pack of A tiles into LDS (fragment layout).
  // Strip s: row r = s>>4, k-range c0 = (s&15)*8 .. +8. A-fragment mapping:
  //   lanes 0-15 : j 0..7 <-> K kb+0..7,   j 8..15 <-> K kb+8..15
  //   lanes 16-31: j 0..7 <-> K kb+16..23, j 8..15 <-> K kb+24..31
  // => lane = r + ((kk>>4)&1)*16 ; dword base = ((kk>>3)&1)*4   (kk = c0&31)
  for (int s = tid; s < 256; s += (OUTD / 16) * 32) {
    const int r  = s >> 4;
    const int c0 = (s & 15) * 8;
    const int kt = c0 >> 5;
    const int kk = c0 & 31;
    const int ln = r + ((kk >> 4) & 1) * 16;
    const int dw = ((kk >> 3) & 1) * 4;
    const float ic = 1.0f / fmaxf(cnt[rbase + r], 1.0f);  // fused mean
    const float4 a0 = *(const float4*)(agg + (size_t)(rbase + r) * DH + c0);
    const float4 a1 = *(const float4*)(agg + (size_t)(rbase + r) * DH + c0 + 4);
    const float4 x0 = *(const float4*)(xin + (size_t)(rbase + r) * DH + c0);
    const float4 x1 = *(const float4*)(xin + (size_t)(rbase + r) * DH + c0 + 4);
    unsigned int pa[4], px[4];
    pa[0] = pack2bf(a0.x * ic, a0.y * ic);
    pa[1] = pack2bf(a0.z * ic, a0.w * ic);
    pa[2] = pack2bf(a1.x * ic, a1.y * ic);
    pa[3] = pack2bf(a1.z * ic, a1.w * ic);
    px[0] = pack2bf(x0.x, x0.y);
    px[1] = pack2bf(x0.z, x0.w);
    px[2] = pack2bf(x1.x, x1.y);
    px[3] = pack2bf(x1.z, x1.w);
    *(uint4*)&aS[kt][ln][dw] = *(uint4*)pa;
    *(uint4*)&xS[kt][ln][dw] = *(uint4*)px;
  }
  __syncthreads();

  // ---- Stage 2: WMMA main loop
  const float bv = bias[col];
  v8f acc = {bv, bv, bv, bv, bv, bv, bv, bv};

  union Frag { v16bf v; uint4 q[2]; };
#pragma unroll
  for (int kt = 0; kt < 4; ++kt) {
    Frag af, xf, bl, br;
    af.q[0] = *(const uint4*)&aS[kt][lane][0];
    af.q[1] = *(const uint4*)&aS[kt][lane][4];
    xf.q[0] = *(const uint4*)&xS[kt][lane][0];
    xf.q[1] = *(const uint4*)&xS[kt][lane][4];
    const size_t fb = (size_t)((wave * 4 + kt) * 32 + lane) * 2;
    bl.q[0] = Wlp[fb];
    bl.q[1] = Wlp[fb + 1];
    br.q[0] = Wrp[fb];
    br.q[1] = Wrp[fb + 1];
    acc = __builtin_amdgcn_wmma_f32_16x16x32_bf16(
        false, af.v, false, bl.v, (short)0, acc, false, false);
    acc = __builtin_amdgcn_wmma_f32_16x16x32_bf16(
        false, xf.v, false, br.v, (short)0, acc, false, false);
  }

  if constexpr (!DO_LN) {
    // C/D layout: VGPR r -> M = r + half*8, N = col
#pragma unroll
    for (int r = 0; r < 8; ++r)
      out[(size_t)(rbase + r + half * 8) * OUTD + col] = acc[r];
  } else {
#pragma unroll
    for (int r = 0; r < 8; ++r)
      tile[r + half * 8][col] = acc[r];
    __syncthreads();

    const int rr  = tid >> 4;   // 256 threads when OUTD==128
    const int seg = tid & 15;
    float s = 0.f, q = 0.f;
#pragma unroll
    for (int c = seg * 8; c < seg * 8 + 8; ++c) {
      const float v = tile[rr][c];
      s += v;
      q += v * v;
    }
    psum[rr][seg] = s;
    psq[rr][seg]  = q;
    __syncthreads();

    if (tid < 16) {
      float ts = 0.f, tq = 0.f;
#pragma unroll
      for (int j = 0; j < 16; ++j) { ts += psum[tid][j]; tq += psq[tid][j]; }
      const float mu  = ts * (1.0f / DH);
      const float var = tq * (1.0f / DH) - mu * mu;
      mu_s[tid] = mu;
      rs_s[tid] = rsqrtf(var + LN_EPS);
    }
    __syncthreads();

    const float mu = mu_s[rr];
    const float rs = rs_s[rr];
#pragma unroll
    for (int c = seg * 8; c < seg * 8 + 8; ++c) {
      const float v = fmaf((tile[rr][c] - mu) * rs, gamma[c], beta[c]);
      out[(size_t)(rbase + rr) * DH + c] = fmaxf(v, 0.f);
    }
  }
}

// ---------------------------------------------------------------------------
extern "C" void kernel_launch(void* const* d_in, const int* in_sizes, int n_in,
                              void* d_out, int out_size, void* d_ws,
                              size_t ws_size, hipStream_t stream) {
  (void)in_sizes; (void)n_in; (void)out_size; (void)ws_size;

  const float*     x     = (const float*)d_in[0];
  const long long* ei    = (const long long*)d_in[1];   // int64 [2,E]
  const float*     Wl1   = (const float*)d_in[2];
  const float*     Wr1   = (const float*)d_in[3];
  const float*     b1    = (const float*)d_in[4];
  const float*     Wl2   = (const float*)d_in[5];
  const float*     Wr2   = (const float*)d_in[6];
  const float*     b2    = (const float*)d_in[7];
  const float*     Wl3   = (const float*)d_in[8];
  const float*     Wr3   = (const float*)d_in[9];
  const float*     b3    = (const float*)d_in[10];
  const float*     gamma = (const float*)d_in[11];
  const float*     beta  = (const float*)d_in[12];
  float*           outp  = (float*)d_out;

  const long long* srcE = ei;
  const long long* dstE = ei + NEDGES;

  // workspace layout: cnt | agg | h1 | h2 | packed weights
  float* cnt = (float*)d_ws;              // NNODES
  float* agg = cnt + NNODES;              // NNODES*DH
  float* h1  = agg + (size_t)NNODES * DH; // NNODES*DH
  float* h2  = h1  + (size_t)NNODES * DH; // NNODES*DH
  unsigned int* wp   = (unsigned int*)(h2 + (size_t)NNODES * DH);
  const int FR128 = (DH / 16) * 4 * 32 * 8;     // dwords per 128-col matrix
  const int FR64  = (DOUT3 / 16) * 4 * 32 * 8;  // dwords per 64-col matrix
  unsigned int* wl1p = wp;
  unsigned int* wr1p = wl1p + FR128;
  unsigned int* wl2p = wr1p + FR128;
  unsigned int* wr2p = wl2p + FR128;
  unsigned int* wl3p = wr2p + FR128;
  unsigned int* wr3p = wl3p + FR64;

  const int nAgg   = NNODES * DH;
  const int zAll4  = (NNODES + nAgg) / 4;
  const int zAgg4  = nAgg / 4;
  const int zb     = 256;
  const int scatB  = (NEDGES * 32) / 256;   // wave per edge, 8 waves/block
  const int rtiles = NNODES / 16;           // 2500

  // one-time prep: weight packing + degree counts + zero cnt/agg
  k_zero4<<<(zAll4 + zb - 1) / zb, zb, 0, stream>>>((float4*)cnt, zAll4);
  k_pack_w<DH><<<4, 256, 0, stream>>>(Wl1, wl1p);
  k_pack_w<DH><<<4, 256, 0, stream>>>(Wr1, wr1p);
  k_pack_w<DH><<<4, 256, 0, stream>>>(Wl2, wl2p);
  k_pack_w<DH><<<4, 256, 0, stream>>>(Wr2, wr2p);
  k_pack_w<DOUT3><<<2, 256, 0, stream>>>(Wl3, wl3p);
  k_pack_w<DOUT3><<<2, 256, 0, stream>>>(Wr3, wr3p);
  k_degree<<<(NEDGES + zb - 1) / zb, zb, 0, stream>>>(dstE, cnt, NEDGES);

  // layer 1
  k_scatter<<<scatB, zb, 0, stream>>>(x, srcE, dstE, agg, NEDGES);
  k_sage_gemm<DH, true><<<rtiles, 256, 0, stream>>>(
      agg, cnt, x, (const uint4*)wl1p, (const uint4*)wr1p, b1, gamma, beta, h1);

  // layer 2
  k_zero4<<<(zAgg4 + zb - 1) / zb, zb, 0, stream>>>((float4*)agg, zAgg4);
  k_scatter<<<scatB, zb, 0, stream>>>(h1, srcE, dstE, agg, NEDGES);
  k_sage_gemm<DH, true><<<rtiles, 256, 0, stream>>>(
      agg, cnt, h1, (const uint4*)wl2p, (const uint4*)wr2p, b2, gamma, beta, h2);

  // layer 3 (no LN / ReLU), 4 column tiles -> 128 threads
  k_zero4<<<(zAgg4 + zb - 1) / zb, zb, 0, stream>>>((float4*)agg, zAgg4);
  k_scatter<<<scatB, zb, 0, stream>>>(h2, srcE, dstE, agg, NEDGES);
  k_sage_gemm<DOUT3, false><<<rtiles, 128, 0, stream>>>(
      agg, cnt, h2, (const uint4*)wl3p, (const uint4*)wr3p, b3, gamma, beta,
      outp);
}
